// SelfAttention_38689065402658
// MI455X (gfx1250) — matrix operand order repaired and checked by reading, not compiled, not run on hardware
//
#include <hip/hip_runtime.h>
#include <hip/hip_bf16.h>

// ---------- types ----------
typedef _Float16 h8   __attribute__((ext_vector_type(8)));
typedef _Float16 v16h __attribute__((ext_vector_type(16)));
typedef float    v8f  __attribute__((ext_vector_type(8)));

__device__ __forceinline__ v16h combine(h8 lo, h8 hi) {
  v16h r;
#pragma unroll
  for (int i = 0; i < 8; ++i) { r[i] = lo[i]; r[i + 8] = hi[i]; }
  return r;
}

// =====================================================================
// Tiled GEMM: C[M,N] = A[M,K] * B[K,N] (+bias), f16 WMMA, f32 acc.
// Block tile 128x128, 256 threads (8 waves), each wave 32x64 (2x4 frags):
// per K-step a wave issues 6 fragment loads for 8 WMMAs.
// =====================================================================
template <typename AT, typename OT, bool BIAS>
__global__ __launch_bounds__(256) void gemm_wmma(
    const AT* __restrict__ A, const float* __restrict__ B,
    const float* __restrict__ bias, OT* __restrict__ Cout,
    int M, int N, int K) {
  constexpr int AS = 40;  // LDS row stride (halves), 32 used, mult of 8
  constexpr int BS = 40;
  __shared__ _Float16 As[128 * AS];
  __shared__ _Float16 Bt[128 * BS];  // transposed: Bt[n][k]

  const int tid  = threadIdx.x;
  const int lane = tid & 31;
  const int wave = tid >> 5;     // 0..7
  const int wm = wave & 3;       // 4 m-slots of 32 rows
  const int wn = wave >> 2;      // 2 n-slots of 64 cols
  const int l16 = lane & 15, half = lane >> 4;
  const int bm = blockIdx.y * 128;
  const int bn = blockIdx.x * 128;

  v8f acc[2][4] = {};

  for (int k0 = 0; k0 < K; k0 += 32) {
    __syncthreads();
    // stage A tile (128x32), convert -> f16
#pragma unroll
    for (int i = 0; i < 16; ++i) {
      int e = i * 256 + tid;            // 0..4095
      int r = e >> 5, c = e & 31;
      As[r * AS + c] = (_Float16)A[(size_t)(bm + r) * K + k0 + c];
    }
    // stage B tile (32x128) transposed; contiguous n -> coalesced reads
#pragma unroll
    for (int i = 0; i < 16; ++i) {
      int e = i * 256 + tid;
      int c = e >> 7, n = e & 127;
      Bt[n * BS + c] = (_Float16)B[(size_t)(k0 + c) * N + bn + n];
    }
    __syncthreads();

    v16h af[2], bf[4];
#pragma unroll
    for (int mt = 0; mt < 2; ++mt) {
      const _Float16* p = &As[(wm * 32 + mt * 16 + l16) * AS + half * 8];
      af[mt] = combine(*(const h8*)p, *(const h8*)(p + 16));
    }
#pragma unroll
    for (int nt = 0; nt < 4; ++nt) {
      const _Float16* p = &Bt[(wn * 64 + nt * 16 + l16) * BS + half * 8];
      bf[nt] = combine(*(const h8*)p, *(const h8*)(p + 16));
    }
#pragma unroll
    for (int mt = 0; mt < 2; ++mt)
#pragma unroll
      for (int nt = 0; nt < 4; ++nt)
        acc[mt][nt] = __builtin_amdgcn_wmma_f32_16x16x32_f16(
            false, af[mt], false, bf[nt], (short)0, acc[mt][nt], false, false);
  }

  // write: D layout -> lane = N + (M/8)*16, reg = M%8
#pragma unroll
  for (int mt = 0; mt < 2; ++mt)
#pragma unroll
    for (int nt = 0; nt < 4; ++nt) {
      int n = bn + wn * 64 + nt * 16 + l16;
#pragma unroll
      for (int r = 0; r < 8; ++r) {
        int m = bm + wm * 32 + mt * 16 + r + half * 8;
        float v = acc[mt][nt][r];
        if constexpr (BIAS) v += bias[n];
        Cout[(size_t)m * N + n] = (OT)v;
      }
    }
}

// =====================================================================
// Flash-style causal attention, f16 WMMA, f32 online softmax.
// qkv: (B*T, 3C) f16; Q at col h*64, K at C+h*64, V at 2C+h*64.
// Block: (q-tile 64, h, b), 128 threads, wave w owns q rows [w*16,w*16+16).
// Scale (1/sqrt(hd)) folded into Q fragments; interior kv-tiles skip the
// causal compare (only the diagonal tile runs it).
// =====================================================================
__global__ __launch_bounds__(128) void attn_wmma(
    const _Float16* __restrict__ qkv, const int* __restrict__ pad,
    _Float16* __restrict__ yout) {
  constexpr int T = 2048, HD = 64, C = 1024, C3 = 3072;
  constexpr int VS = 72;  // Vt row stride (halves)
  constexpr int PS = 72;
  __shared__ _Float16 Vt[64 * VS];     // V^T: [hd][key]
  __shared__ _Float16 Pw[4][16 * PS];  // per-wave P strips

  const int tid  = threadIdx.x;
  const int lane = tid & 31, wave = tid >> 5;
  const int l16 = lane & 15, half = lane >> 4;
  const int qt = blockIdx.x, h = blockIdx.y, b = blockIdx.z;
  const int qbase = qt * 64;
  const size_t rowbase = (size_t)b * T;

  // Q A-fragments (2 hd chunks), loaded once, pre-scaled by 1/sqrt(64)
  v16h qf[2];
  {
    const int qrow = qbase + wave * 16 + l16;
    const _Float16* qp = qkv + (rowbase + qrow) * C3 + h * HD;
#pragma unroll
    for (int kc = 0; kc < 2; ++kc) {
      const _Float16* p = qp + kc * 32 + half * 8;
      v16h q = combine(*(const h8*)p, *(const h8*)(p + 16));
#pragma unroll
      for (int i = 0; i < 16; ++i) q[i] = q[i] * (_Float16)0.125f;
      qf[kc] = q;
    }
  }

  float mrow[8], lrow[8];
#pragma unroll
  for (int r = 0; r < 8; ++r) { mrow[r] = -1e30f; lrow[r] = 0.f; }
  v8f o[4] = {};

  for (int kt = 0; kt <= qt; ++kt) {
    const int kbase = kt * 64;

    // stage V^T tile
    __syncthreads();
#pragma unroll
    for (int i = 0; i < 32; ++i) {
      int e = i * 128 + tid;          // 0..4095
      int key = e >> 6, d = e & 63;   // contiguous d -> coalesced
      Vt[d * VS + key] = qkv[(rowbase + kbase + key) * C3 + 2 * C + h * HD + d];
    }
    __syncthreads();

    // S = Q * K^T  (K^T B-fragments load directly from row-major K)
    v8f s[4];
#pragma unroll
    for (int nt = 0; nt < 4; ++nt) {
      v8f z = {};
#pragma unroll
      for (int kc = 0; kc < 2; ++kc) {
        const _Float16* kp = qkv + (rowbase + kbase + nt * 16 + l16) * C3 +
                             C + h * HD + kc * 32 + half * 8;
        v16h kf = combine(*(const h8*)kp, *(const h8*)(kp + 16));
        z = __builtin_amdgcn_wmma_f32_16x16x32_f16(false, qf[kc], false, kf,
                                                   (short)0, z, false, false);
      }
      s[nt] = z;
    }

    // mask + row max (causal compare only on the diagonal tile)
    float lm[8];
#pragma unroll
    for (int r = 0; r < 8; ++r) lm[r] = -1e30f;
    if (kt == qt) {
#pragma unroll
      for (int nt = 0; nt < 4; ++nt) {
        int key = kbase + nt * 16 + l16;
        int pv = pad[rowbase + key];
#pragma unroll
        for (int r = 0; r < 8; ++r) {
          int q = qbase + wave * 16 + r + half * 8;
          float v = s[nt][r];
          if (key > q || pv == 0) v = -1e30f;
          s[nt][r] = v;
          lm[r] = fmaxf(lm[r], v);
        }
      }
    } else {
#pragma unroll
      for (int nt = 0; nt < 4; ++nt) {
        int key = kbase + nt * 16 + l16;
        int pv = pad[rowbase + key];
#pragma unroll
        for (int r = 0; r < 8; ++r) {
          float v = s[nt][r];
          if (pv == 0) v = -1e30f;
          s[nt][r] = v;
          lm[r] = fmaxf(lm[r], v);
        }
      }
    }
#pragma unroll
    for (int r = 0; r < 8; ++r) {
#pragma unroll
      for (int off = 1; off < 16; off <<= 1)
        lm[r] = fmaxf(lm[r], __shfl_xor(lm[r], off, 32));
    }

    // online-softmax update
    float alpha[8], psum[8];
#pragma unroll
    for (int r = 0; r < 8; ++r) {
      float mnew = fmaxf(mrow[r], lm[r]);
      alpha[r] = __expf(mrow[r] - mnew);
      mrow[r] = mnew;
      psum[r] = 0.f;
    }
    _Float16* pp = Pw[wave];
#pragma unroll
    for (int nt = 0; nt < 4; ++nt) {
#pragma unroll
      for (int r = 0; r < 8; ++r) {
        float p = __expf(s[nt][r] - mrow[r]);
        psum[r] += p;
        pp[(r + half * 8) * PS + nt * 16 + l16] = (_Float16)p;  // D -> LDS
      }
    }
#pragma unroll
    for (int r = 0; r < 8; ++r) {
      float t = psum[r];
#pragma unroll
      for (int off = 1; off < 16; off <<= 1) t += __shfl_xor(t, off, 32);
      lrow[r] = lrow[r] * alpha[r] + t;
    }
#pragma unroll
    for (int nt = 0; nt < 4; ++nt)
#pragma unroll
      for (int r = 0; r < 8; ++r) o[nt][r] *= alpha[r];

    // reload P as A-fragments (same-wave LDS RAW is in-order)
    v16h pf[2];
#pragma unroll
    for (int kc = 0; kc < 2; ++kc) {
      const _Float16* p = &pp[l16 * PS + kc * 32 + half * 8];
      pf[kc] = combine(*(const h8*)p, *(const h8*)(p + 16));
    }
    // O += P * V
#pragma unroll
    for (int nt = 0; nt < 4; ++nt) {
#pragma unroll
      for (int kc = 0; kc < 2; ++kc) {
        const _Float16* vp = &Vt[(nt * 16 + l16) * VS + kc * 32 + half * 8];
        v16h vf = combine(*(const h8*)vp, *(const h8*)(vp + 16));
        o[nt] = __builtin_amdgcn_wmma_f32_16x16x32_f16(false, pf[kc], false, vf,
                                                       (short)0, o[nt], false, false);
      }
    }
  }

  // normalize, write y (B*T, C) f16 at col h*64+d
#pragma unroll
  for (int nt = 0; nt < 4; ++nt) {
#pragma unroll
    for (int r = 0; r < 8; ++r) {
      int m = qbase + wave * 16 + r + half * 8;
      float v = o[nt][r] / lrow[r];
      yout[(rowbase + m) * C + h * HD + nt * 16 + l16] = (_Float16)v;
    }
  }
}

// =====================================================================
// launcher
// =====================================================================
extern "C" void kernel_launch(void* const* d_in, const int* in_sizes, int n_in,
                              void* d_out, int out_size, void* d_ws, size_t ws_size,
                              hipStream_t stream) {
  (void)in_sizes; (void)n_in; (void)out_size; (void)ws_size;
  const float* x      = (const float*)d_in[0];
  const int*   pad    = (const int*)d_in[1];
  const float* w_attn = (const float*)d_in[2];
  const float* w_proj = (const float*)d_in[3];
  const float* b_proj = (const float*)d_in[4];
  float* out = (float*)d_out;

  constexpr int Bb = 4, T = 2048, C = 1024, H = 16;
  constexpr int M = Bb * T;  // 8192
  _Float16* qkv   = (_Float16*)d_ws;                                   // 48 MB
  _Float16* yattn = (_Float16*)((char*)d_ws + (size_t)M * 3 * C * 2);  // 16 MB

  // 1) qkv = x @ w_attn
  gemm_wmma<float, _Float16, false>
      <<<dim3(3 * C / 128, M / 128), dim3(256), 0, stream>>>(
          x, w_attn, nullptr, qkv, M, 3 * C, C);
  // 2) attention
  attn_wmma<<<dim3(T / 64, H, Bb), dim3(128), 0, stream>>>(qkv, pad, yattn);
  // 3) out = y @ w_proj + b_proj
  gemm_wmma<_Float16, float, true>
      <<<dim3(C / 128, M / 128), dim3(256), 0, stream>>>(
          yattn, w_proj, b_proj, out, M, C, C);
}
